// DEINA_JBF_67714454389313
// MI455X (gfx1250) — compile-verified, matrix operand order checked by reading, not run on hardware
//
#include <hip/hip_runtime.h>
#include <hip/hip_bf16.h>

typedef __attribute__((ext_vector_type(16))) __bf16 v16bf;
typedef __attribute__((ext_vector_type(8)))  __bf16 v8bf;
typedef __attribute__((ext_vector_type(8)))  float  v8f;

#define BUFS 264   // 256 + 8 halfs pad -> 528B row stride, conflict-free ds_load_b128

__device__ inline float leaky(float v){ return v > 0.f ? v : 0.01f * v; }

__device__ inline void atomicAddF32(float* p, float v){
  __hip_atomic_fetch_add(p, v, __ATOMIC_RELAXED, __HIP_MEMORY_SCOPE_AGENT);
}

// ---------------------------------------------------------------------------
// Stats: per-feature mean/var (training-mode BN) for xs [4096,16] and us [4032,8]
// stats layout: [0..15]=xmu [16..31]=xvar [32..39]=umu [40..47]=uvar
// ---------------------------------------------------------------------------
__global__ void k_stats(const float* __restrict__ xs, const float* __restrict__ us,
                        float* __restrict__ stats){
  __shared__ float s1[256], s2[256];
  int tid = threadIdx.x;
  const float* src; int M, D; float* mu; float* var;
  if (blockIdx.x == 0){ src = xs; M = 4096; D = 16; mu = stats;      var = stats + 16; }
  else                { src = us; M = 4032; D = 8;  mu = stats + 32; var = stats + 40; }
  int perF = 256 / D;
  int f = tid % D, g = tid / D;
  float a = 0.f, b = 0.f;
  for (int r = g; r < M; r += perF){ float v = src[r*D + f]; a += v; b += v*v; }
  s1[tid] = a; s2[tid] = b;
  __syncthreads();
  for (int h = 128; h >= D; h >>= 1){
    if (tid < h){ s1[tid] += s1[tid + h]; s2[tid] += s2[tid + h]; }
    __syncthreads();
  }
  if (tid < D){ float m = s1[tid] / (float)M; mu[tid] = m; var[tid] = s2[tid] / (float)M - m*m; }
}

// ---------------------------------------------------------------------------
// Weight repack: f32 [Kd][N] (per net) -> bf16 WMMA B-fragments (32x16 tiles),
// per-lane-contiguous 16 halfs.  Also folds gamma into W1 and computes
// b1eff = b1 + beta @ W1.
// ---------------------------------------------------------------------------
__device__ inline void pack_lane(__bf16* out, const float* __restrict__ W,
                                 const float* __restrict__ gamma,
                                 int Kd, int N, int kt, int nt, int lane){
  int n  = nt*16 + (lane & 15);
  int kb = kt*32 + ((lane >> 4) << 3);
#pragma unroll
  for (int i = 0; i < 16; i++){
    int kk = kb + ((i < 8) ? i : (i + 8));   // halfs 0..7 -> K kb..kb+7, 8..15 -> kb+16..kb+23
    float v = 0.f;
    if (kk < Kd){ v = W[(size_t)kk*N + n]; if (gamma) v *= gamma[kk]; }
    out[i] = (__bf16)v;
  }
}

__global__ void k_repack(const float* __restrict__ xW1, const float* __restrict__ xW2,
                         const float* __restrict__ xW3, const float* __restrict__ x_gamma,
                         const float* __restrict__ x_beta, const float* __restrict__ xb1,
                         const float* __restrict__ uW1, const float* __restrict__ uW2,
                         const float* __restrict__ uW3, const float* __restrict__ u_gamma,
                         const float* __restrict__ u_beta, const float* __restrict__ ub1,
                         __bf16* __restrict__ wb1x, __bf16* __restrict__ wb2x, __bf16* __restrict__ wb3x,
                         __bf16* __restrict__ wb1u, __bf16* __restrict__ wb2u, __bf16* __restrict__ wb3u,
                         float* __restrict__ beffx, float* __restrict__ beffu){
  long idx = (long)blockIdx.x * 256 + threadIdx.x;
  const long n1 = 32768, n2 = 262144, n3 = 131072;   // lanes per region
  if (idx < n1){  // xW1: 1 ktile (16 pad 32) x 16 ntiles
    long li = idx; int net = (int)(li >> 9); int rem = (int)(li & 511);
    pack_lane(wb1x + li*16, xW1 + (size_t)net*16*256, x_gamma + net*16, 16, 256, 0, rem >> 5, rem & 31);
    return;
  }
  idx -= n1;
  if (idx < n2){  // xW2: 8 kt x 16 nt
    long li = idx; int net = (int)(li >> 12); int rem = (int)(li & 4095);
    int tile = rem >> 5;
    pack_lane(wb2x + li*16, xW2 + (size_t)net*65536, nullptr, 256, 256, tile >> 4, tile & 15, rem & 31);
    return;
  }
  idx -= n2;
  if (idx < n3){  // xW3: 8 kt x 8 nt
    long li = idx; int net = (int)(li >> 11); int rem = (int)(li & 2047);
    int tile = rem >> 5;
    pack_lane(wb3x + li*16, xW3 + (size_t)net*32768, nullptr, 256, 128, tile >> 3, tile & 7, rem & 31);
    return;
  }
  idx -= n3;
  if (idx < n1){  // uW1: Kd=8 pad 32
    long li = idx; int net = (int)(li >> 9); int rem = (int)(li & 511);
    pack_lane(wb1u + li*16, uW1 + (size_t)net*8*256, u_gamma + net*8, 8, 256, 0, rem >> 5, rem & 31);
    return;
  }
  idx -= n1;
  if (idx < n2){
    long li = idx; int net = (int)(li >> 12); int rem = (int)(li & 4095);
    int tile = rem >> 5;
    pack_lane(wb2u + li*16, uW2 + (size_t)net*65536, nullptr, 256, 256, tile >> 4, tile & 15, rem & 31);
    return;
  }
  idx -= n2;
  if (idx < n3){
    long li = idx; int net = (int)(li >> 11); int rem = (int)(li & 2047);
    int tile = rem >> 5;
    pack_lane(wb3u + li*16, uW3 + (size_t)net*32768, nullptr, 256, 128, tile >> 3, tile & 7, rem & 31);
    return;
  }
  idx -= n3;
  if (idx < 16384){  // beffx: b1 + beta @ W1 (folded BN shift)
    int t = (int)idx; int k = t >> 8, h = t & 255;
    float acc = xb1[t];
    for (int d = 0; d < 16; d++) acc += x_beta[k*16 + d] * xW1[((size_t)k*16 + d)*256 + h];
    beffx[t] = acc;
    return;
  }
  idx -= 16384;
  if (idx < 16384){
    int t = (int)idx; int k = t >> 8, h = t & 255;
    float acc = ub1[t];
    for (int d = 0; d < 8; d++) acc += u_beta[k*8 + d] * uW1[((size_t)k*8 + d)*256 + h];
    beffu[t] = acc;
  }
}

// ---------------------------------------------------------------------------
// Build shared (net-independent) normalized-input A-fragments, bf16,
// per-lane contiguous (K padded 16->32 / 8->32 with zeros).
// ---------------------------------------------------------------------------
__global__ void k_frag(const float* __restrict__ xs, const float* __restrict__ us,
                       const float* __restrict__ stats,
                       __bf16* __restrict__ xnfrag, __bf16* __restrict__ unfrag){
  int gid = blockIdx.x*256 + threadIdx.x;  // x: 256 tiles*32 lanes, u: 252*32
  if (gid < 8192){
    int tile = gid >> 5, lane = gid & 31;
    int m = tile*16 + (lane & 15);
    int kb = (lane >> 4) << 3;
    __bf16* dst = xnfrag + (size_t)gid*16;
#pragma unroll
    for (int i = 0; i < 16; i++){
      int kk = kb + ((i < 8) ? i : (i + 8));
      float v = 0.f;
      if (kk < 16) v = (xs[m*16 + kk] - stats[kk]) * rsqrtf(stats[16 + kk] + 1e-5f);
      dst[i] = (__bf16)v;
    }
  } else if (gid < 8192 + 8064){
    int g2 = gid - 8192;
    int tile = g2 >> 5, lane = g2 & 31;
    int m = tile*16 + (lane & 15);
    int kb = (lane >> 4) << 3;
    __bf16* dst = unfrag + (size_t)g2*16;
#pragma unroll
    for (int i = 0; i < 16; i++){
      int kk = kb + ((i < 8) ? i : (i + 8));
      float v = 0.f;
      if (kk < 8) v = (us[m*8 + kk] - stats[32 + kk]) * rsqrtf(stats[40 + kk] + 1e-5f);
      dst[i] = (__bf16)v;
    }
  }
}

// ---------------------------------------------------------------------------
// psi(0): BN(0) = 0 -> affine gives beta -> layer1 out = b1eff. Tiny per-net MLP.
// ---------------------------------------------------------------------------
__global__ void k_psi0(const float* __restrict__ uW2, const float* __restrict__ ub2,
                       const float* __restrict__ uW3, const float* __restrict__ u_scale,
                       const float* __restrict__ beffu, float* __restrict__ psi0){
  int k = blockIdx.x, tid = threadIdx.x;
  __shared__ float h1[256], h2[256];
  h1[tid] = leaky(beffu[k*256 + tid]);
  __syncthreads();
  float acc = ub2[k*256 + tid];
  const float* W2 = uW2 + (size_t)k*65536;
  for (int d = 0; d < 256; d++) acc += h1[d] * W2[(size_t)d*256 + tid];
  h2[tid] = leaky(acc);
  __syncthreads();
  if (tid < 128){
    float a = 0.f;
    const float* W3 = uW3 + (size_t)k*32768;
    for (int d = 0; d < 256; d++) a += h2[d] * W3[(size_t)d*128 + tid];
    psi0[k*128 + tid] = a * u_scale[k*128 + tid];
  }
}

// ---------------------------------------------------------------------------
// Fused 3-layer MLP per (net, 64-row block) using v_wmma_f32_16x16x32_bf16.
// 8 waves: wave -> (mt = w&3, half = w>>2) tile ownership. LDS ping-pong bufs.
// mode 0 (x): scale, atomic-add into phisum, store per-net t==0 rows to phi0.
// mode 1 (u): scale, subtract psi0, store bf16 per-net ures.
// ---------------------------------------------------------------------------
__device__ inline v16bf load_afrag_lds(const __bf16* buf, int mt, int kt, int lane){
  int row = mt*16 + (lane & 15);
  int kb  = kt*32 + ((lane >> 4) << 3);
  const __bf16* p = buf + row*BUFS + kb;
  v8bf lo = *(const v8bf*)p;
  v8bf hi = *(const v8bf*)(p + 16);
  return __builtin_shufflevector(lo, hi, 0,1,2,3,4,5,6,7,8,9,10,11,12,13,14,15);
}

__global__ __launch_bounds__(256) void k_mlp(
    const __bf16* __restrict__ afrag,
    const __bf16* __restrict__ wb1, const __bf16* __restrict__ wb2, const __bf16* __restrict__ wb3,
    const float* __restrict__ beff1, const float* __restrict__ b2, const float* __restrict__ scale,
    const float* __restrict__ psi0,
    float* __restrict__ phisum, float* __restrict__ phi0, __bf16* __restrict__ ures,
    int M, int mode){
  __shared__ __bf16 buf1[64*BUFS];
  __shared__ __bf16 buf2[64*BUFS];
  int rbCount = M >> 6;
  int net = blockIdx.x / rbCount;
  int rb  = blockIdx.x % rbCount;
  int tid  = threadIdx.x;
  int lane = tid & 31;
  int wave = tid >> 5;
  int mt   = wave & 3;     // 4 row tiles of 16
  int half = wave >> 2;    // column-range split

  // ---- Layer 1: A from global fragments (shared across nets), Kd padded to 32 ----
  {
    v16bf a = *(const v16bf*)(afrag + (((size_t)(rb*4 + mt))*32 + lane)*16);
    const __bf16* wnet = wb1 + (size_t)net*16*512;
#pragma unroll
    for (int q = 0; q < 8; q++){
      int nt = half*8 + q;
      v16bf b = *(const v16bf*)(wnet + ((size_t)nt*32 + lane)*16);
      v8f acc = {};
      acc = __builtin_amdgcn_wmma_f32_16x16x32_bf16(false, a, false, b, (short)0, acc, false, false);
      int col  = nt*16 + (lane & 15);
      float bs = beff1[net*256 + col];
      int row0 = mt*16 + ((lane >> 4) << 3);
#pragma unroll
      for (int r = 0; r < 8; r++)
        buf1[(row0 + r)*BUFS + col] = (__bf16)leaky(acc[r] + bs);
    }
  }
  __syncthreads();

  // ---- Layer 2: 256->256, 8 K-tiles, A fragments cached in VGPRs ----
  {
    v16bf a[8];
#pragma unroll
    for (int kt = 0; kt < 8; kt++) a[kt] = load_afrag_lds(buf1, mt, kt, lane);
    const __bf16* wnet = wb2 + (size_t)net*128*512;
#pragma unroll
    for (int q = 0; q < 8; q++){
      int nt = half*8 + q;
      v8f acc = {};
#pragma unroll
      for (int kt = 0; kt < 8; kt++){
        v16bf b = *(const v16bf*)(wnet + (((size_t)kt*16 + nt)*32 + lane)*16);
        acc = __builtin_amdgcn_wmma_f32_16x16x32_bf16(false, a[kt], false, b, (short)0, acc, false, false);
      }
      int col  = nt*16 + (lane & 15);
      float bs = b2[net*256 + col];
      int row0 = mt*16 + ((lane >> 4) << 3);
#pragma unroll
      for (int r = 0; r < 8; r++)
        buf2[(row0 + r)*BUFS + col] = (__bf16)leaky(acc[r] + bs);
    }
  }
  __syncthreads();

  // ---- Layer 3: 256->128, no bias, output scale + finalize ----
  {
    v16bf a[8];
#pragma unroll
    for (int kt = 0; kt < 8; kt++) a[kt] = load_afrag_lds(buf2, mt, kt, lane);
    const __bf16* wnet = wb3 + (size_t)net*64*512;
#pragma unroll
    for (int q = 0; q < 4; q++){
      int nt = half*4 + q;
      v8f acc = {};
#pragma unroll
      for (int kt = 0; kt < 8; kt++){
        v16bf b = *(const v16bf*)(wnet + (((size_t)kt*8 + nt)*32 + lane)*16);
        acc = __builtin_amdgcn_wmma_f32_16x16x32_bf16(false, a[kt], false, b, (short)0, acc, false, false);
      }
      int col  = nt*16 + (lane & 15);
      float sc = scale[net*128 + col];
      int row0 = rb*64 + mt*16 + ((lane >> 4) << 3);
      if (mode == 0){
#pragma unroll
        for (int r = 0; r < 8; r++){
          int m = row0 + r;
          float v = acc[r] * sc;
          atomicAddF32(&phisum[(size_t)m*128 + col], v);           // sum over nets (y path)
          if ((m & 63) == 0)                                       // t == 0: recurrence init, per net
            phi0[((size_t)net*64 + (m >> 6))*128 + col] = v;
        }
      } else {
        float p0 = psi0[net*128 + col];
#pragma unroll
        for (int r = 0; r < 8; r++){
          int m = row0 + r;
          ures[((size_t)net*4032 + m)*128 + col] = (__bf16)(acc[r] * sc - p0);
        }
      }
    }
  }
}

// ---------------------------------------------------------------------------
// Recurrence: one block per batch row b. State p[net][128] in LDS.
// p <- (p + u_t) R_j per net; y_pred[b,t] = (sum_j p_j) @ C^T fused in.
// ---------------------------------------------------------------------------
__global__ __launch_bounds__(256) void k_recur(const float* __restrict__ phi0,
    const __bf16* __restrict__ ures, const float* __restrict__ reL,
    const float* __restrict__ imL, const float* __restrict__ CW,
    float* __restrict__ ypred){
  int b = blockIdx.x, tid = threadIdx.x;
  __shared__ float p[64*128];
  __shared__ float ssum[128];
  __shared__ float re_s[64], im_s[64];
  __shared__ float cw_s[2048];
  for (int idx = tid; idx < 8192; idx += 256){
    int j = idx >> 7, col = idx & 127;
    p[idx] = phi0[((size_t)j*64 + b)*128 + col];
  }
  for (int idx = tid; idx < 2048; idx += 256) cw_s[idx] = CW[idx];
  if (tid < 64){ re_s[tid] = reL[tid]; im_s[tid] = imL[tid]; }
  __syncthreads();
  for (int t = 0; t < 63; t++){
    for (int idx = tid; idx < 4096; idx += 256){   // (net j, pair pr)
      int j = idx >> 6, pr = idx & 63;
      const __bf16* up = ures + ((size_t)j*4032 + b*63 + t)*128 + pr*2;
      float a0 = p[j*128 + 2*pr]     + (float)up[0];
      float a1 = p[j*128 + 2*pr + 1] + (float)up[1];
      float r = re_s[j], i = im_s[j];
      p[j*128 + 2*pr]     = a0*r - a1*i;
      p[j*128 + 2*pr + 1] = a0*i + a1*r;
    }
    __syncthreads();
    if (tid < 128){
      float s = 0.f;
      for (int j = 0; j < 64; j++) s += p[j*128 + tid];
      ssum[tid] = s;
    }
    __syncthreads();
    if (tid < 16){
      float acc = 0.f;
      for (int l = 0; l < 128; l++) acc += ssum[l] * cw_s[tid*128 + l];
      ypred[(b*63 + t)*16 + tid] = acc;
    }
    __syncthreads();
  }
}

// y path decode: y[b,t'] = phisum[b*64 + t' + 1] @ C^T
__global__ void k_decode(const float* __restrict__ phisum, const float* __restrict__ CW,
                         float* __restrict__ y){
  int idx = blockIdx.x*256 + threadIdx.x;
  if (idx >= 64512) return;
  int rrow = idx >> 4, d = idx & 15;
  int b = rrow / 63, t = rrow % 63;
  const float* pr = phisum + ((size_t)b*64 + t + 1)*128;
  const float* cw = CW + d*128;
  float acc = 0.f;
  for (int l = 0; l < 128; l++) acc += pr[l] * cw[l];
  y[idx] = acc;
}

// ---------------------------------------------------------------------------
extern "C" void kernel_launch(void* const* d_in, const int* in_sizes, int n_in,
                              void* d_out, int out_size, void* d_ws, size_t ws_size,
                              hipStream_t stream){
  const float* xs      = (const float*)d_in[0];
  const float* us      = (const float*)d_in[1];
  const float* x_gamma = (const float*)d_in[2];
  const float* x_beta  = (const float*)d_in[3];
  const float* xW1     = (const float*)d_in[4];
  const float* xb1     = (const float*)d_in[5];
  const float* xW2     = (const float*)d_in[6];
  const float* xb2     = (const float*)d_in[7];
  const float* xW3     = (const float*)d_in[8];
  const float* x_scale = (const float*)d_in[9];
  const float* u_gamma = (const float*)d_in[10];
  const float* u_beta  = (const float*)d_in[11];
  const float* uW1     = (const float*)d_in[12];
  const float* ub1     = (const float*)d_in[13];
  const float* uW2     = (const float*)d_in[14];
  const float* ub2     = (const float*)d_in[15];
  const float* uW3     = (const float*)d_in[16];
  const float* u_scale = (const float*)d_in[17];
  const float* reL     = (const float*)d_in[18];
  const float* imL     = (const float*)d_in[19];
  const float* C_W     = (const float*)d_in[20];

  char* ws = (char*)d_ws;
  size_t off = 0;
  auto take = [&](size_t bytes) -> char* {
    char* p = ws + off;
    off += (bytes + 255) & ~(size_t)255;
    return p;
  };
  float*  stats  = (float*) take(48u * 4);
  float*  phisum = (float*) take((size_t)4096*128*4);
  float*  phi0   = (float*) take((size_t)64*64*128*4);
  float*  psi0   = (float*) take((size_t)64*128*4);
  float*  beffx  = (float*) take((size_t)64*256*4);
  float*  beffu  = (float*) take((size_t)64*256*4);
  __bf16* xnfrag = (__bf16*)take((size_t)8192*16*2);
  __bf16* unfrag = (__bf16*)take((size_t)8064*16*2);
  __bf16* wb1x   = (__bf16*)take((size_t)64*16*512*2);
  __bf16* wb2x   = (__bf16*)take((size_t)64*128*512*2);
  __bf16* wb3x   = (__bf16*)take((size_t)64*64*512*2);
  __bf16* wb1u   = (__bf16*)take((size_t)64*16*512*2);
  __bf16* wb2u   = (__bf16*)take((size_t)64*128*512*2);
  __bf16* wb3u   = (__bf16*)take((size_t)64*64*512*2);
  __bf16* ures   = (__bf16*)take((size_t)64*4032*128*2);

  hipMemsetAsync(phisum, 0, (size_t)4096*128*4, stream);
  k_stats<<<2, 256, 0, stream>>>(xs, us, stats);
  k_repack<<<3456, 256, 0, stream>>>(xW1, xW2, xW3, x_gamma, x_beta, xb1,
                                     uW1, uW2, uW3, u_gamma, u_beta, ub1,
                                     wb1x, wb2x, wb3x, wb1u, wb2u, wb3u, beffx, beffu);
  k_frag<<<64, 256, 0, stream>>>(xs, us, stats, xnfrag, unfrag);
  k_psi0<<<64, 256, 0, stream>>>(uW2, ub2, uW3, u_scale, beffu, psi0);
  k_mlp<<<4096, 256, 0, stream>>>(xnfrag, wb1x, wb2x, wb3x, beffx, xb2, x_scale,
                                  nullptr, phisum, phi0, nullptr, 4096, 0);
  k_mlp<<<4032, 256, 0, stream>>>(unfrag, wb1u, wb2u, wb3u, beffu, ub2, u_scale,
                                  psi0, nullptr, nullptr, ures, 4032, 1);
  float* y     = (float*)d_out;
  float* ypred = y + 64512;
  k_recur<<<64, 256, 0, stream>>>(phi0, ures, reL, imL, C_W, ypred);
  k_decode<<<252, 256, 0, stream>>>(phisum, C_W, y);
}
